// Conv2d_21148418966220
// MI455X (gfx1250) — matrix operand order repaired and checked by reading, not compiled, not run on hardware
//
#include <hip/hip_runtime.h>

typedef __bf16 v2bf  __attribute__((ext_vector_type(2)));
typedef __bf16 v8bf  __attribute__((ext_vector_type(8)));
typedef __bf16 v16bf __attribute__((ext_vector_type(16)));
typedef float  v8f   __attribute__((ext_vector_type(8)));

#define C_IN   128
#define H_IN   56
#define W_IN   56
#define K_OC   256
#define HW     (H_IN * W_IN)          // 3136
#define N_IMG  32
#define NPIX   (N_IMG * HW)           // 100352
#define WSTR   (C_IN * 9)             // 1152: weight oc stride (OIHW)

#define M_BLK  128                    // oc per block
#define N_BLK  128                    // pixels per block
#define KC     32                     // channels per stage (= one bf16 WMMA K)
#define NSTAGE 36                     // 9 taps * 4 channel chunks
#define PITCH  40                     // bf16 per LDS row (32 data + 8 pad; 80B rows)

// Permute channel position within an A row so each lane's A fragment
// (16-bit A 16x32 layout: lanes 0-15 need K={0..7}u{16..23}, lanes 16-31 need
// K={8..15}u{24..31}) is one contiguous 32B run. Keeps even/odd pairs adjacent.
__device__ __forceinline__ int a_perm(int c) {
  return (c & 7) | ((c & 8) << 1) | ((c & 16) >> 1);
}

__global__ __launch_bounds__(256, 2) void conv2d_bf16x3_wmma(
    const float* __restrict__ in, const float* __restrict__ wt,
    const float* __restrict__ bias, float* __restrict__ out)
{
  __shared__ __align__(16) __bf16 As_hi[M_BLK * PITCH];  // [oc][perm(c)]
  __shared__ __align__(16) __bf16 As_lo[M_BLK * PITCH];
  __shared__ __align__(16) __bf16 Bs_hi[N_BLK * PITCH];  // [px][perm(c)]
  __shared__ __align__(16) __bf16 Bs_lo[N_BLK * PITCH];

  const int tid  = threadIdx.x;
  const int lane = tid & 31;
  const int wave = tid >> 5;
  const int wm   = wave & 1;         // 0..1 : 64-oc half of block tile
  const int wn   = wave >> 1;        // 0..3 : 32-px column of block tile

  const int oc_blk = blockIdx.y * M_BLK;
  const int px_blk = blockIdx.x * N_BLK;

  // ---- staging maps (each thread owns 16 consecutive channels) ----
  // B: 128 px x 32 c. thread -> (px = tid&127, c in [(tid>>7)*16, +16))
  const int pxl  = tid & 127;
  const int cb0  = (tid >> 7) * 16;
  const int pxg  = px_blk + pxl;
  const int nB   = pxg / HW;
  const int remB = pxg - nB * HW;
  const int yB   = remB / W_IN;
  const int xB   = remB - yB * W_IN;
  const int inBase = nB * (C_IN * HW);
  // A: 128 oc x 32 c. thread -> (oc = tid>>1, c in [(tid&1)*16, +16))
  const int ocA = tid >> 1;
  const int cg  = (tid & 1) * 16;

  v8f acc[4][2];
  #pragma unroll
  for (int t = 0; t < 4; ++t)
    #pragma unroll
    for (int u = 0; u < 2; ++u) acc[t][u] = (v8f){0,0,0,0,0,0,0,0};

  const int m    = lane & 15;
  const int half = lane >> 4;

  float aReg[16], bReg[16];

  // ---------- pipeline: load stage kt+1 to regs while computing stage kt ----------
  // stage kt: rs = kt>>2 (tap), c0 = (kt&3)*32 (channel chunk)
  #define LOAD_STAGE(KT)                                                        \
    {                                                                           \
      const int rs = (KT) >> 2;                                                 \
      const int c0 = ((KT) & 3) << 5;                                           \
      const int r  = rs / 3, s = rs - r * 3;                                    \
      _Pragma("unroll")                                                         \
      for (int i = 0; i < 16; ++i)                                              \
        aReg[i] = wt[(oc_blk + ocA) * WSTR + (c0 + cg + i) * 9 + rs];           \
      const int yy = yB + r - 1, xx = xB + s - 1;                               \
      const bool ok = (yy >= 0) && (yy < H_IN) && (xx >= 0) && (xx < W_IN);     \
      const int off = inBase + yy * W_IN + xx;                                  \
      _Pragma("unroll")                                                         \
      for (int j = 0; j < 16; ++j)                                              \
        bReg[j] = ok ? in[off + (c0 + cb0 + j) * HW] : 0.0f;                    \
    }

  #define STORE_STAGE()                                                         \
    {                                                                           \
      _Pragma("unroll")                                                         \
      for (int i = 0; i < 16; i += 2) {                                         \
        const int pos = a_perm(cg + i);                                         \
        const v2bf hh = { (__bf16)aReg[i], (__bf16)aReg[i + 1] };               \
        const v2bf ll = { (__bf16)(aReg[i]     - (float)hh.x),                  \
                          (__bf16)(aReg[i + 1] - (float)hh.y) };                \
        *(v2bf*)&As_hi[ocA * PITCH + pos] = hh;                                 \
        *(v2bf*)&As_lo[ocA * PITCH + pos] = ll;                                 \
      }                                                                         \
      _Pragma("unroll")                                                         \
      for (int j = 0; j < 16; j += 2) {                                         \
        const int c = a_perm(cb0 + j);                                          \
        const v2bf hh = { (__bf16)bReg[j], (__bf16)bReg[j + 1] };               \
        const v2bf ll = { (__bf16)(bReg[j]     - (float)hh.x),                  \
                          (__bf16)(bReg[j + 1] - (float)hh.y) };                \
        *(v2bf*)&Bs_hi[pxl * PITCH + c] = hh;                                   \
        *(v2bf*)&Bs_lo[pxl * PITCH + c] = ll;                                   \
      }                                                                         \
    }

  LOAD_STAGE(0);
  STORE_STAGE();

  #pragma unroll 1
  for (int kt = 0; kt < NSTAGE; ++kt) {
    __syncthreads();                       // staged LDS visible to all waves
    if (kt + 1 < NSTAGE) LOAD_STAGE(kt + 1);   // overlap next-stage globals with WMMA

    // ---- compute: 24 WMMAs (4 M-subtiles x 2 N-subtiles x bf16x3 split) ----
    v16bf bh[2], bl[2];
    #pragma unroll
    for (int u = 0; u < 2; ++u) {
      const int brow = wn * 32 + u * 16 + m;
      const v8bf* pbh = (const v8bf*)&Bs_hi[brow * PITCH + half * 16];
      const v8bf* pbl = (const v8bf*)&Bs_lo[brow * PITCH + half * 16];
      bh[u] = __builtin_shufflevector(pbh[0], pbh[1],
              0,1,2,3,4,5,6,7,8,9,10,11,12,13,14,15);
      bl[u] = __builtin_shufflevector(pbl[0], pbl[1],
              0,1,2,3,4,5,6,7,8,9,10,11,12,13,14,15);
    }
    #pragma unroll
    for (int t = 0; t < 4; ++t) {
      const int arow = wm * 64 + t * 16 + m;
      const v8bf* pah = (const v8bf*)&As_hi[arow * PITCH + half * 16];
      const v8bf* pal = (const v8bf*)&As_lo[arow * PITCH + half * 16];
      const v16bf ah = __builtin_shufflevector(pah[0], pah[1],
              0,1,2,3,4,5,6,7,8,9,10,11,12,13,14,15);
      const v16bf al = __builtin_shufflevector(pal[0], pal[1],
              0,1,2,3,4,5,6,7,8,9,10,11,12,13,14,15);
      #pragma unroll
      for (int u = 0; u < 2; ++u) {
        acc[t][u] = __builtin_amdgcn_wmma_f32_16x16x32_bf16(
            false, ah, false, bh[u], (short)0, acc[t][u], false, false);
        acc[t][u] = __builtin_amdgcn_wmma_f32_16x16x32_bf16(
            false, ah, false, bl[u], (short)0, acc[t][u], false, false);
        acc[t][u] = __builtin_amdgcn_wmma_f32_16x16x32_bf16(
            false, al, false, bh[u], (short)0, acc[t][u], false, false);
      }
    }

    __syncthreads();                       // all waves done reading this stage
    if (kt + 1 < NSTAGE) STORE_STAGE();    // convert + commit next stage to LDS
  }

  // ---- epilogue: C/D layout (VGPR v: lanes0-15 M=v, lanes16-31 M=v+8) + bias ----
  #pragma unroll
  for (int u = 0; u < 2; ++u) {
    const int pxo  = px_blk + wn * 32 + u * 16 + m;
    const int nO   = pxo / HW;
    const int remO = pxo - nO * HW;
    const int outPix = nO * (K_OC * HW) + remO;
    #pragma unroll
    for (int t = 0; t < 4; ++t) {
      #pragma unroll
      for (int v = 0; v < 8; ++v) {
        const int oc = oc_blk + wm * 64 + t * 16 + half * 8 + v;
        out[outPix + oc * HW] = acc[t][u][v] + bias[oc];
      }
    }
  }
}

extern "C" void kernel_launch(void* const* d_in, const int* in_sizes, int n_in,
                              void* d_out, int out_size, void* d_ws, size_t ws_size,
                              hipStream_t stream) {
  const float* in   = (const float*)d_in[0];   // 32x128x56x56 f32
  const float* wt   = (const float*)d_in[1];   // 256x128x3x3 f32
  const float* bias = (const float*)d_in[2];   // 256 f32
  float* out = (float*)d_out;                  // 32x256x56x56 f32

  dim3 grid(NPIX / N_BLK, K_OC / M_BLK, 1);    // 784 x 2
  conv2d_bf16x3_wmma<<<grid, 256, 0, stream>>>(in, wt, bias, out);
}
